// TheanoGRU_39178691674954
// MI455X (gfx1250) — matrix operand order repaired and checked by reading, not compiled, not run on hardware
//
#include <hip/hip_runtime.h>
#include <hip/hip_bf16.h>

typedef __attribute__((ext_vector_type(16))) __bf16 v16bf;
typedef __attribute__((ext_vector_type(4)))  __bf16 v4bf;
typedef __attribute__((ext_vector_type(8)))  float  v8f;
typedef __attribute__((ext_vector_type(4)))  float  f4;

// Global (addrspace 1) pointer types so laundered pointers still lower to
// global_load_* (flat_load would also consume DScnt and the LDS pipe).
typedef __attribute__((address_space(1))) const __bf16 gbf;
typedef __attribute__((address_space(1))) const f4     gf4;

namespace {
constexpr int kB = 128;
constexpr int kT = 1024;
constexpr int kD = 256;
constexpr int kH = 256;
constexpr int HP = 264;              // padded LDS pitch (elems): 4-bank rotation/row, 16B aligned
constexpr int MATE = kD * kH;        // 65536 elems per packed matrix
}

struct F32x8 { f4 lo, hi; };

__device__ __forceinline__ v16bf frag_from(f4 lo, f4 hi) {
  F32x8 t{lo, hi};
  return __builtin_bit_cast(v16bf, t);
}

// A-fragment: 16 bf16 per lane as two 16B chunks at +0 elems and +16 elems (K, K+16)
__device__ __forceinline__ v16bf load_a2(const __bf16* p) {
  return frag_from(((const f4*)p)[0], ((const f4*)(p + 16))[0]);
}

// B-fragment: 16 contiguous bf16 per lane from the packed-weight layout (global)
__device__ __forceinline__ v16bf load_b32B(gbf* p) {
  gf4* p4 = (gf4*)p;
  return frag_from(p4[0], p4[1]);
}

__device__ __forceinline__ v8f wmma_bf16(v16bf a, v16bf b, v8f c) {
  return __builtin_amdgcn_wmma_f32_16x16x32_bf16(false, a, false, b, (short)0, c,
                                                 false, false);
}

__device__ __forceinline__ float hsig(float x) {
  return fminf(fmaxf(0.2f * x + 0.5f, 0.0f), 1.0f);
}

// ---------------------------------------------------------------------------
// Kernel 1: x fp32 -> bf16 (vectorized)
// ---------------------------------------------------------------------------
__global__ void cvt_x_kernel(const float* __restrict__ src,
                             __bf16* __restrict__ dst, int n4) {
  int i = blockIdx.x * blockDim.x + threadIdx.x;
  const int stride = gridDim.x * blockDim.x;
  const f4* s4 = (const f4*)src;
  v4bf* d4 = (v4bf*)dst;
  for (; i < n4; i += stride) {
    f4 v = s4[i];
    v4bf o;
    o[0] = (__bf16)v[0]; o[1] = (__bf16)v[1];
    o[2] = (__bf16)v[2]; o[3] = (__bf16)v[3];
    d4[i] = o;
  }
}

// ---------------------------------------------------------------------------
// Kernel 2: pack 256x256 fp32 row-major weights into WMMA B-operand layout.
// Per (n_tile, k_tile) 32x16 block (1 KB): lane l, dword j holds the bf16 pair
//   K = k_tile*32 + (l<16?0:8) + 2j + (j<4?0:8) (+1),  N = n_tile*16 + (l%16)
// stored contiguously per lane (32 B/lane).
// ---------------------------------------------------------------------------
__global__ void pack_w_kernel(const float* __restrict__ w0, const float* __restrict__ w1,
                              const float* __restrict__ w2, const float* __restrict__ w3,
                              const float* __restrict__ w4, const float* __restrict__ w5,
                              __bf16* __restrict__ dst) {
  const float* srcs[6] = {w0, w1, w2, w3, w4, w5};
  const float* s = srcs[blockIdx.x];
  __bf16* d = dst + (size_t)blockIdx.x * MATE;
  for (int idx = threadIdx.x; idx < MATE / 2; idx += blockDim.x) {
    const int blk = idx >> 8;          // 128 blocks of 256 pairs
    const int within = idx & 255;
    const int lane = within >> 3;
    const int j = within & 7;
    const int nt = blk >> 3, kt = blk & 7;
    const int k = kt * 32 + ((lane < 16) ? 0 : 8) + 2 * j + ((j < 4) ? 0 : 8);
    const int n = nt * 16 + (lane & 15);
    d[2 * idx + 0] = (__bf16)s[(size_t)k * kH + n];
    d[2 * idx + 1] = (__bf16)s[(size_t)(k + 1) * kH + n];
  }
}

// ---------------------------------------------------------------------------
// Kernel 3: persistent GRU scan. 8 WGs (batch tiles of 16), 8 waves each.
// Wave w owns N-tiles {2w, 2w+1}. Per step:
//   phase A: z,r = hsig(x_t@W + h@U + b)   (64 WMMAs/wave, fused W+U)
//   phase B: hh = tanh(x_t@W_h + (r*h)@U_h + b_h); h = z*h + (1-z)*hh
// h kept in LDS (fp32 + bf16 A-layout copy); weights streamed from WGP$/L2 in
// the packed B layout. The weight base pointer is laundered through an
// asm volatile each timestep (as a global-addrspace pointer) so LLVM cannot
// hoist the loop-invariant fragment loads out of the t-loop and spill them,
// while still emitting global_load_b128 (not flat_load).
// ---------------------------------------------------------------------------
__global__ __launch_bounds__(256, 1)
void gru_scan_kernel(const __bf16* __restrict__ xb,
                     const __bf16* __restrict__ wp,     // [Wz Wr Wh Uz Ur Uh]
                     const float* __restrict__ b_z,
                     const float* __restrict__ b_r,
                     const float* __restrict__ b_h,
                     float* __restrict__ out) {
  __shared__ float  hs[16 * HP];
  __shared__ __bf16 hb[16 * HP];
  __shared__ __bf16 rhb[16 * HP];

  const int tid  = threadIdx.x;
  const int lane = tid & 31;
  const int wave = tid >> 5;
  const int b0   = blockIdx.x * 16;

  for (int i = tid; i < 16 * HP; i += 256) {
    hs[i] = 0.0f; hb[i] = (__bf16)0.0f; rhb[i] = (__bf16)0.0f;
  }
  __syncthreads();

  const int row = lane & 15;   // A-fragment row
  const int hi  = lane >> 4;   // lane-half selector
  const int col = lane & 15;   // C-tile column within tile

  int   ncol[2];
  float bzv[2], brv[2], bhv[2];
#pragma unroll
  for (int q = 0; q < 2; ++q) {
    const int n = wave * 32 + q * 16 + col;
    ncol[q] = n;
    bzv[q] = b_z[n]; brv[q] = b_r[n]; bhv[q] = b_h[n];
  }

  const __bf16* xrow0 = xb + (size_t)(b0 + row) * kT * kD;

  float zsv[2][8], hsv[2][8];

#pragma unroll 1
  for (int t = 0; t < kT; ++t) {
    const __bf16* xrow = xrow0 + (size_t)t * kD;

    // Launder the weight base pointer (global addrspace): keeps the
    // loop-invariant fragment loads inside the loop as global_load_b128,
    // reading from WGP$/L2 instead of spilling 96 fragments to scratch.
    gbf* wpt = (gbf*)wp;
    asm volatile("" : "+s"(wpt));
    gbf* Wz = wpt + 0 * (size_t)MATE;
    gbf* Wr = wpt + 1 * (size_t)MATE;
    gbf* Wh = wpt + 2 * (size_t)MATE;
    gbf* Uz = wpt + 3 * (size_t)MATE;
    gbf* Ur = wpt + 4 * (size_t)MATE;
    gbf* Uh = wpt + 5 * (size_t)MATE;

    // ----- phase A: gates z and r -------------------------------------
    v8f accz[2] = {};
    v8f accr[2] = {};
#pragma unroll
    for (int kt = 0; kt < 8; ++kt) {
      const int aoff = kt * 32 + hi * 8;
      v16bf ax = load_a2(xrow + aoff);
      v16bf ah = load_a2(&hb[row * HP + aoff]);
#pragma unroll
      for (int q = 0; q < 2; ++q) {
        const int nt = wave * 2 + q;
        const int boff = (nt * 8 + kt) * 512 + lane * 16;
        accz[q] = wmma_bf16(ax, load_b32B(Wz + boff), accz[q]);
        accz[q] = wmma_bf16(ah, load_b32B(Uz + boff), accz[q]);
        accr[q] = wmma_bf16(ax, load_b32B(Wr + boff), accr[q]);
        accr[q] = wmma_bf16(ah, load_b32B(Ur + boff), accr[q]);
      }
    }
#pragma unroll
    for (int q = 0; q < 2; ++q) {
#pragma unroll
      for (int i = 0; i < 8; ++i) {
        const int m = hi * 8 + i;
        const float z  = hsig(accz[q][i] + bzv[q]);
        const float r  = hsig(accr[q][i] + brv[q]);
        const float h0 = hs[m * HP + ncol[q]];
        zsv[q][i] = z;
        hsv[q][i] = h0;
        rhb[m * HP + ncol[q]] = (__bf16)(r * h0);
      }
    }
    __syncthreads();

    // ----- phase B: candidate + state update --------------------------
    v8f acch[2] = {};
#pragma unroll
    for (int kt = 0; kt < 8; ++kt) {
      const int aoff = kt * 32 + hi * 8;
      v16bf ax = load_a2(xrow + aoff);
      v16bf ar = load_a2(&rhb[row * HP + aoff]);
#pragma unroll
      for (int q = 0; q < 2; ++q) {
        const int nt = wave * 2 + q;
        const int boff = (nt * 8 + kt) * 512 + lane * 16;
        acch[q] = wmma_bf16(ax, load_b32B(Wh + boff), acch[q]);
        acch[q] = wmma_bf16(ar, load_b32B(Uh + boff), acch[q]);
      }
    }
    if (t + 1 < kT) __builtin_prefetch(xrow + kD, 0, 1);  // next timestep's x row
#pragma unroll
    for (int q = 0; q < 2; ++q) {
#pragma unroll
      for (int i = 0; i < 8; ++i) {
        const int m = hi * 8 + i;
        const float hh = tanhf(acch[q][i] + bhv[q]);
        const float hn = zsv[q][i] * hsv[q][i] + (1.0f - zsv[q][i]) * hh;
        hs[m * HP + ncol[q]] = hn;
        hb[m * HP + ncol[q]] = (__bf16)hn;
      }
    }
    __syncthreads();
  }

  for (int i = tid; i < 16 * kH; i += 256) {
    const int r = i >> 8, c = i & 255;
    out[(size_t)(b0 + r) * kH + c] = hs[r * HP + c];
  }
}

// ---------------------------------------------------------------------------
extern "C" void kernel_launch(void* const* d_in, const int* in_sizes, int n_in,
                              void* d_out, int out_size, void* d_ws, size_t ws_size,
                              hipStream_t stream) {
  (void)in_sizes; (void)n_in; (void)out_size; (void)ws_size;
  const float* x   = (const float*)d_in[0];
  const float* W_z = (const float*)d_in[1];
  const float* W_r = (const float*)d_in[2];
  const float* W_h = (const float*)d_in[3];
  const float* U_z = (const float*)d_in[4];
  const float* U_r = (const float*)d_in[5];
  const float* U_h = (const float*)d_in[6];
  const float* b_z = (const float*)d_in[7];
  const float* b_r = (const float*)d_in[8];
  const float* b_h = (const float*)d_in[9];
  float* out = (float*)d_out;

  // workspace: [ x bf16 : 64 MiB ][ 6 packed bf16 weights : 768 KiB ]
  __bf16* xb = (__bf16*)d_ws;
  __bf16* wp = (__bf16*)((char*)d_ws + (size_t)kB * kT * kD * sizeof(__bf16));

  cvt_x_kernel<<<2048, 256, 0, stream>>>(x, xb, kB * kT * kD / 4);
  pack_w_kernel<<<6, 256, 0, stream>>>(W_z, W_r, W_h, U_z, U_r, U_h, wp);
  gru_scan_kernel<<<kB / 16, 256, 0, stream>>>(xb, wp, b_z, b_r, b_h, out);
}
